// AttentiveQuantizer_71708773974330
// MI455X (gfx1250) — compile-verified, matrix operand order checked by reading, not run on hardware
//
#include <hip/hip_runtime.h>

// Problem constants
#define NB   8
#define CIN  192
#define COUT 192
#define HH   64
#define WW   64
#define KC   1024
#define P    (NB*HH*WW)        // 32768 positions
#define PLANE (HH*WW)          // 4096

// ---- vector types (trivial, union-safe) ----
typedef __attribute__((ext_vector_type(16))) __bf16       v16bf;
typedef __attribute__((ext_vector_type(8)))  float        v8f;
typedef __attribute__((ext_vector_type(4)))  unsigned int u32x4;

union FragBF { u32x4 q[2]; v16bf v; };

__device__ __forceinline__ unsigned short f2bf(float f) {
    unsigned int x = __float_as_uint(f);
    unsigned int r = x + 0x7FFFu + ((x >> 16) & 1u);   // round-to-nearest-even
    return (unsigned short)(r >> 16);
}

__device__ __forceinline__ float urand01(unsigned int x) {
    // PCG-style hash -> uniform (0,1], never 0
    x = x * 747796405u + 2891336453u;
    unsigned int w = ((x >> ((x >> 28) + 4u)) ^ x) * 277803737u;
    w = (w >> 22) ^ w;
    return (float)(w >> 8) * (1.0f / 16777216.0f) + (0.5f / 16777216.0f);
}

// CDNA5 async memory->LDS copy, 16 bytes per lane, tracked by ASYNCcnt.
// GV addressing mode: per-lane 32-bit LDS dest VGPR + 64-bit global addr VGPR pair.
__device__ __forceinline__ void async_copy16(unsigned lds_addr, const void* gaddr) {
    asm volatile("global_load_async_to_lds_b128 %0, %1, off"
                 :: "v"(lds_addr), "v"(gaddr) : "memory");
}

// ---------------------------------------------------------------------------
// Kernel 1: k = codebook @ wk^T (bf16), v = codebook @ wv^T (f32), wq -> bf16
// Tiny (150 MFLOP) -> scalar is fine; everything L2-resident.
// ---------------------------------------------------------------------------
__global__ __launch_bounds__(256) void prep_kernel(
    const float* __restrict__ codebook, const float* __restrict__ wk,
    const float* __restrict__ wv, const float* __restrict__ wq,
    unsigned short* __restrict__ kmatb, float* __restrict__ vmat,
    unsigned short* __restrict__ wqb)
{
    int t = blockIdx.x * 256 + threadIdx.x;          // < KC*COUT = 196608
    if (t < COUT * CIN) wqb[t] = f2bf(wq[t]);
    int kk = t / COUT, o = t % COUT;
    const float* crow = codebook + kk * COUT;
    const float* krow = wk + o * COUT;
    const float* vrow = wv + o * COUT;
    float dk = 0.f, dv = 0.f;
    #pragma unroll 8
    for (int c = 0; c < COUT; ++c) { float cb = crow[c]; dk += cb * krow[c]; dv += cb * vrow[c]; }
    kmatb[t] = f2bf(dk);
    vmat[t]  = dv;
}

// ---------------------------------------------------------------------------
// Kernel 2: q[p,o] = sum_c latent[n,c,h,w]*wq[o,c]  -> q_ws (bf16, row-major)
// Per block: 128 positions. Latent tile transposed into LDS as bf16 (needs
// f32->bf16 conversion, so it must pass through VGPRs), B fragments read
// straight from bf16 wq in global (73 KB, L2-resident).
// ---------------------------------------------------------------------------
__global__ __launch_bounds__(256) void qgemm_kernel(
    const float* __restrict__ latent, const unsigned short* __restrict__ wqb,
    unsigned short* __restrict__ qws)
{
    __shared__ unsigned short ls[128 * 200];   // [p_local][c], padded stride
    const int tid   = threadIdx.x;
    const int ptile = blockIdx.x * 128;
    const int n     = ptile >> 12;             // tile never crosses image boundary
    const int s0    = ptile & (PLANE - 1);
    const float* lbase = latent + (size_t)n * CIN * PLANE + s0;

    // stage: CIN x 128 f32 -> bf16, transposed (coalesced global reads along p)
    for (int j = 0; j < (CIN * 128) / 256; ++j) {
        int lin = j * 256 + tid;
        int c = lin >> 7;            // /128
        int p = lin & 127;
        ls[p * 200 + c] = f2bf(lbase[c * PLANE + p]);
    }
    __syncthreads();

    const int lane = tid & 31, w = tid >> 5;
    const int hi   = (lane & 16) ? 8 : 0;      // 16-bit A/B fragment K-half select
    const int arow = w * 16 + (lane & 15);
    const int mhalf = (lane & 16) ? 8 : 0;

    FragBF af[6];
    #pragma unroll
    for (int k6 = 0; k6 < 6; ++k6) {
        int kb = k6 * 32;
        af[k6].q[0] = *(const u32x4*)(ls + arow * 200 + kb + hi);
        af[k6].q[1] = *(const u32x4*)(ls + arow * 200 + kb + 16 + hi);
    }
    for (int nt = 0; nt < COUT / 16; ++nt) {
        v8f acc = {0.f,0.f,0.f,0.f,0.f,0.f,0.f,0.f};
        int orow = nt * 16 + (lane & 15);
        #pragma unroll
        for (int k6 = 0; k6 < 6; ++k6) {
            int kb = k6 * 32;
            FragBF bfr;
            bfr.q[0] = *(const u32x4*)(wqb + orow * CIN + kb + hi);
            bfr.q[1] = *(const u32x4*)(wqb + orow * CIN + kb + 16 + hi);
            acc = __builtin_amdgcn_wmma_f32_16x16x32_bf16(
                      false, af[k6].v, false, bfr.v, (short)0, acc, false, false);
        }
        #pragma unroll
        for (int r = 0; r < 8; ++r) {
            int m = w * 16 + mhalf + r;
            qws[(ptile + m) * COUT + nt * 16 + (lane & 15)] = f2bf(acc[r]);
        }
    }
}

// ---------------------------------------------------------------------------
// Kernel 3: logit = q.k^T * t1/sqrt(CIN); fused argmax + gumbel-argmax.
// 128 positions/block; 1024 codes in 16 chunks of 64, double-buffered through
// LDS with GLOBAL_LOAD_ASYNC_TO_LDS_B128 (ASYNCcnt) so staging of chunk cc+1
// overlaps WMMA + epilogue of chunk cc. Async loads complete in order, so
// s_wait_asynccnt 6 (= the just-issued next chunk) guarantees chunk cc landed.
// ---------------------------------------------------------------------------
__global__ __launch_bounds__(256) void logit_kernel(
    const unsigned short* __restrict__ qws, const unsigned short* __restrict__ kmatb,
    const float* __restrict__ t1p, const int* __restrict__ tempp,
    float* __restrict__ logit_out, float* __restrict__ truecode_out,
    int* __restrict__ idxws)
{
    __shared__ unsigned short qs[128 * 200];       // 51.2 KB
    __shared__ unsigned short ks[2][64 * 200];     // 51.2 KB (double buffer)
    __shared__ float rmaxS[128]; __shared__ int ridxS[128];
    __shared__ float gmaxS[128]; __shared__ int gidxS[128];

    const int tid = threadIdx.x;
    const int ptile = blockIdx.x * 128;

    // async-stage q tile: 128 rows x 192 bf16 = 128 x 24 x 16B  (12 issues/thread)
    #pragma unroll
    for (int j = 0; j < 12; ++j) {
        int lin = j * 256 + tid;
        int row = lin / 24;
        int col = (lin % 24) * 8;
        async_copy16((unsigned)(uintptr_t)(qs + row * 200 + col),
                     qws + (ptile + row) * COUT + col);
    }
    // async-stage k chunk 0: 64 rows x 24 x 16B  (6 issues/thread)
    #pragma unroll
    for (int j = 0; j < 6; ++j) {
        int lin = j * 256 + tid;
        int row = lin / 24;
        int col = (lin % 24) * 8;
        async_copy16((unsigned)(uintptr_t)(&ks[0][row * 200 + col]),
                     kmatb + row * COUT + col);
    }
    if (tid < 128) { rmaxS[tid] = -3.4e38f; gmaxS[tid] = -3.4e38f; ridxS[tid] = 0; gidxS[tid] = 0; }

    const int lane = tid & 31, w = tid >> 5;
    const int hi = (lane & 16) ? 8 : 0;
    const int arow = w * 16 + (lane & 15);
    const int mhalf = (lane & 16) ? 8 : 0;
    const float scale = (*t1p) * 0.07216878364870323f;   // 1/sqrt(192)
    const float invT  = 1.0f / (float)(*tempp);

    FragBF af[6];

    for (int cc = 0; cc < 16; ++cc) {
        const int cbase = cc * 64;
        if (cc < 15) {
            // issue next chunk into the other buffer (overlaps this chunk's math)
            #pragma unroll
            for (int j = 0; j < 6; ++j) {
                int lin = j * 256 + tid;
                int row = lin / 24;
                int col = (lin % 24) * 8;
                async_copy16((unsigned)(uintptr_t)(&ks[(cc + 1) & 1][row * 200 + col]),
                             kmatb + (cbase + 64 + row) * COUT + col);
            }
            asm volatile("s_wait_asynccnt 0x6" ::: "memory");  // chunk cc (and qs) done
        } else {
            asm volatile("s_wait_asynccnt 0x0" ::: "memory");
        }
        __syncthreads();

        if (cc == 0) {   // uniform branch: EXEC stays all-ones for WMMA below
            #pragma unroll
            for (int k6 = 0; k6 < 6; ++k6) {
                int kb = k6 * 32;
                af[k6].q[0] = *(const u32x4*)(qs + arow * 200 + kb + hi);
                af[k6].q[1] = *(const u32x4*)(qs + arow * 200 + kb + 16 + hi);
            }
        }

        float bv[4], gv[4]; int bc[4], gc[4];
        // per-m-row running best kept per r; 4 n-subtiles in this chunk
        float bvr[8], gvr[8]; int bcr[8], gcr[8];
        #pragma unroll
        for (int r = 0; r < 8; ++r) { bvr[r] = -3.4e38f; gvr[r] = -3.4e38f; bcr[r] = 0; gcr[r] = 0; }
        (void)bv; (void)gv; (void)bc; (void)gc;

        const unsigned short* ksb = &ks[cc & 1][0];
        for (int t = 0; t < 4; ++t) {
            v8f acc = {0.f,0.f,0.f,0.f,0.f,0.f,0.f,0.f};
            int brow = t * 16 + (lane & 15);
            #pragma unroll
            for (int k6 = 0; k6 < 6; ++k6) {
                int kb = k6 * 32;
                FragBF bfr;
                bfr.q[0] = *(const u32x4*)(ksb + brow * 200 + kb + hi);
                bfr.q[1] = *(const u32x4*)(ksb + brow * 200 + kb + 16 + hi);
                acc = __builtin_amdgcn_wmma_f32_16x16x32_bf16(
                          false, af[k6].v, false, bfr.v, (short)0, acc, false, false);
            }
            const int code = cbase + t * 16 + (lane & 15);
            #pragma unroll
            for (int r = 0; r < 8; ++r) {
                int m = w * 16 + mhalf + r;
                int p = ptile + m;
                float val = acc[r] * scale;
                logit_out[p * KC + code] = val;               // coalesced 64B runs across lanes
                float u = urand01((unsigned)(p * KC + code));
                float g = -logf(-logf(u));
                float pv = (val + g) * invT;
                if (val > bvr[r]) { bvr[r] = val; bcr[r] = code; }
                if (pv  > gvr[r]) { gvr[r] = pv;  gcr[r] = code; }
            }
        }
        // reduce argmax across the 16 lanes holding each row (C/D layout halves)
        #pragma unroll
        for (int r = 0; r < 8; ++r) {
            float v0 = bvr[r]; int c0 = bcr[r];
            float v1 = gvr[r]; int c1 = gcr[r];
            for (int off = 1; off < 16; off <<= 1) {
                float ov = __shfl_xor(v0, off, 32); int oc = __shfl_xor(c0, off, 32);
                if (ov > v0 || (ov == v0 && oc < c0)) { v0 = ov; c0 = oc; }
                float og = __shfl_xor(v1, off, 32); int ogc = __shfl_xor(c1, off, 32);
                if (og > v1 || (og == v1 && ogc < c1)) { v1 = og; c1 = ogc; }
            }
            if ((lane & 15) == 0) {   // lanes 0 and 16 own distinct rows; row owned by one wave
                int m = w * 16 + mhalf + r;
                if (v0 > rmaxS[m]) { rmaxS[m] = v0; ridxS[m] = c0; }
                if (v1 > gmaxS[m]) { gmaxS[m] = v1; gidxS[m] = c1; }
            }
        }
        __syncthreads();   // protect ks[(cc)&1] before it is refilled at cc+1
    }
    if (tid < 128) {
        int p = ptile + tid;
        truecode_out[p] = (float)ridxS[tid];
        idxws[p] = gidxS[tid];
    }
}

// ---------------------------------------------------------------------------
// Kernel 4: quantized[n,c,h,w] = v[idx[p], c]  (hard one-hot == row gather)
// ---------------------------------------------------------------------------
__global__ __launch_bounds__(256) void gather_kernel(
    const float* __restrict__ vmat, const int* __restrict__ idxws,
    float* __restrict__ outq)
{
    int t = blockIdx.x * 256 + threadIdx.x;       // < NB*COUT*PLANE
    int s = t & (PLANE - 1);
    int c = (t >> 12) % COUT;
    int n = t / (COUT * PLANE);
    int p = (n << 12) + s;
    outq[t] = vmat[idxws[p] * COUT + c];          // v is 786KB -> L2 resident
}

// ---------------------------------------------------------------------------
extern "C" void kernel_launch(void* const* d_in, const int* in_sizes, int n_in,
                              void* d_out, int out_size, void* d_ws, size_t ws_size,
                              hipStream_t stream) {
    const float* latent   = (const float*)d_in[0];
    const float* codebook = (const float*)d_in[1];
    const float* wq       = (const float*)d_in[2];
    const float* wk       = (const float*)d_in[3];
    const float* wv       = (const float*)d_in[4];
    const float* t1       = (const float*)d_in[5];
    const int*   temp     = (const int*)d_in[6];

    char* ws = (char*)d_ws;                                     // ~14 MB total
    unsigned short* qws   = (unsigned short*)(ws);              // 32768*192*2  = 12,582,912
    unsigned short* kmatb = (unsigned short*)(ws + 12582912);   // 1024*192*2   =    393,216
    float*          vmat  = (float*)        (ws + 12976128);    // 1024*192*4   =    786,432
    unsigned short* wqb   = (unsigned short*)(ws + 13762560);   // 192*192*2    =     73,728
    int*            idxws = (int*)          (ws + 13836288);    // 32768*4      =    131,072

    float* outq     = (float*)d_out;          // quantized [8,192,64,64]
    float* outtc    = outq + NB * COUT * PLANE;       // trueCode [8,64,64] (as f32)
    float* outlogit = outtc + P;                      // logit [8,64,64,1024]

    prep_kernel  <<<(KC * COUT) / 256, 256, 0, stream>>>(codebook, wk, wv, wq, kmatb, vmat, wqb);
    qgemm_kernel <<<P / 128,           256, 0, stream>>>(latent, wqb, qws);
    logit_kernel <<<P / 128,           256, 0, stream>>>(qws, kmatb, t1, temp, outlogit, outtc, idxws);
    gather_kernel<<<(NB * COUT * PLANE) / 256, 256, 0, stream>>>(vmat, idxws, outq);
}
